// SelfAtten3D_36601711296640
// MI455X (gfx1250) — matrix operand order repaired and checked by reading, not compiled, not run on hardware
//
#include <hip/hip_runtime.h>

// SelfAtten3D on MI455X (gfx1250, wave32, WMMA 16x16x32 f16).
// Flash-attention structure: never materialize the 13824^2 score matrix.
// Round 3: kill the scratch spills introduced by deep load batching:
//  - proj: rolling 4-slot B-fragment pipeline + sched_barrier between chunks
//  - attn: reuse S registers for P, tail-split V double buffer
//
// Workspace layout (needs ~9.02 MB):
//   Q  f16 [N][32]   @ 0
//   K  f16 [N][32]   @ 884736
//   Vt f16 [256][N]  @ 1769472   (V stored transposed for contiguous B-operand loads)
//   Pg f16 panel     @ 8847360   (wg repacked to WMMA B-fragment order)
//   Pf f16 panel     @ 8863744
//   Ph f16 panel     @ 8880128

typedef __attribute__((ext_vector_type(16))) _Float16 v16h;
typedef __attribute__((ext_vector_type(8)))  _Float16 v8h;
typedef __attribute__((ext_vector_type(8)))  float    v8f;

#define NTOK 13824
#define CCH  256

static __device__ __forceinline__ v8f wmma_f16(v16h a, v16h b, v8f c) {
    return __builtin_amdgcn_wmma_f32_16x16x32_f16(false, a, false, b, (short)0, c, false, false);
}

// Repack a f32 weight matrix W[256][ncCount*16] into f16 B-fragment panels:
// P[((kc*ncCount + nc)*16 + n)*32 + kh*16 + e] = W[kc*32 + kh*16 + e][nc*16 + n]
__global__ void pack_panel(const float* __restrict__ W, _Float16* __restrict__ P,
                           int ncCount, int ldW, int total) {
    int i = blockIdx.x * blockDim.x + threadIdx.x;
    if (i >= total) return;
    int e  = i & 15;
    int kh = (i >> 4) & 1;
    int n  = (i >> 5) & 15;
    int nc = (i >> 9) % ncCount;
    int kc = i / (512 * ncCount);
    P[i] = (_Float16)W[(kc * 32 + kh * 16 + e) * ldW + nc * 16 + n];
}

// One wave per 16-token tile: Q = xWg+bg, K = xWf+bf, V = xWh+bh (V stored transposed).
__global__ void proj_kernel(const float* __restrict__ x,
                            const float* __restrict__ bg, const float* __restrict__ bf,
                            const float* __restrict__ bh,
                            const _Float16* __restrict__ Pg, const _Float16* __restrict__ Pf,
                            const _Float16* __restrict__ Ph,
                            _Float16* __restrict__ Q, _Float16* __restrict__ K,
                            _Float16* __restrict__ Vt) {
    const int lane = threadIdx.x & 31;
    const int wid  = threadIdx.x >> 5;
    const int tok0 = (blockIdx.x * 4 + wid) * 16;
    const int col  = lane & 15;
    const int half = lane >> 4;

    // A-fragments of x tile: row = lane&15; elem e -> k = 16*half+e
    v16h Xa[8];
#pragma unroll
    for (int kc = 0; kc < 8; ++kc) {
        const float* p0 = x + (size_t)(tok0 + col) * CCH + kc * 32 + half * 16;
        v8f lo = *(const v8f*)p0;
        v8f hi = *(const v8f*)(p0 + 8);
        v16h a;
#pragma unroll
        for (int e = 0; e < 8; ++e) { a[e] = (_Float16)lo[e]; a[8 + e] = (_Float16)hi[e]; }
        Xa[kc] = a;
    }

    // Rolling 4-slot B pipeline: up to 8 loads in flight, only ~5 fragments live.
    auto gemm_chunk = [&](const _Float16* panel, int ncCount, int nc) -> v8f {
        const _Float16* base = panel + ((size_t)nc * 16 + col) * 32 + half * 16;
        const size_t kstride = (size_t)ncCount * 16 * 32;
        v16h b[4];
#pragma unroll
        for (int kc = 0; kc < 4; ++kc)
            b[kc] = *(const v16h*)(base + kc * kstride);
        v8f acc = {};
#pragma unroll
        for (int kc = 0; kc < 4; ++kc) {
            v16h cur = b[kc];
            b[kc] = *(const v16h*)(base + (size_t)(kc + 4) * kstride);
            acc = wmma_f16(Xa[kc], cur, acc);
        }
#pragma unroll
        for (int kc = 0; kc < 4; ++kc)
            acc = wmma_f16(Xa[kc + 4], b[kc], acc);
        return acc;
    };

    // Q and K: 2 output chunks of 16 each, row-major f16 [N][32].
    // sched_barrier keeps the chunks' live ranges from being merged (spills).
#pragma unroll
    for (int nc = 0; nc < 2; ++nc) {
        v8f aq = gemm_chunk(Pg, 2, nc);
        float bq = bg[nc * 16 + col];
#pragma unroll
        for (int r = 0; r < 8; ++r)
            Q[(size_t)(tok0 + r + 8 * half) * 32 + nc * 16 + col] = (_Float16)(aq[r] + bq);
        __builtin_amdgcn_sched_barrier(0);
        v8f ak = gemm_chunk(Pf, 2, nc);
        float bk = bf[nc * 16 + col];
#pragma unroll
        for (int r = 0; r < 8; ++r)
            K[(size_t)(tok0 + r + 8 * half) * 32 + nc * 16 + col] = (_Float16)(ak[r] + bk);
        __builtin_amdgcn_sched_barrier(0);
    }

    // V: 16 output chunks, stored transposed Vt[d][token]; rows per lane contiguous -> b128 store
#pragma unroll
    for (int nc = 0; nc < 16; ++nc) {
        v8f av = gemm_chunk(Ph, 16, nc);
        float bv = bh[nc * 16 + col];
        v8h pk;
#pragma unroll
        for (int r = 0; r < 8; ++r) pk[r] = (_Float16)(av[r] + bv);
        *(v8h*)(Vt + (size_t)(nc * 16 + col) * NTOK + tok0 + 8 * half) = pk;
        __builtin_amdgcn_sched_barrier(0);
    }
}

// Flash attention: one wave per 16-query tile, key blocks of 32, online softmax,
// O tile 16x256 f32 resident in 128 VGPRs, accumulated with v_wmma_f32_16x16x32_f16.
__global__ void attn_kernel(const float* __restrict__ x, const float* __restrict__ gamma,
                            const _Float16* __restrict__ Q, const _Float16* __restrict__ K,
                            const _Float16* __restrict__ Vt, float* __restrict__ out) {
    __shared__ _Float16 Pbuf[4][16 * 32];   // per-wave D-layout -> A-layout bounce buffer
    const int lane = threadIdx.x & 31;
    const int wid  = threadIdx.x >> 5;
    const int col  = lane & 15;
    const int half = lane >> 4;
    const int tok0 = (blockIdx.x * 4 + wid) * 16;
    _Float16* pb = Pbuf[wid];

    // Q A-fragment (16x32 f16)
    v16h Qa;
    {
        const _Float16* q0 = Q + (size_t)(tok0 + col) * 32 + half * 16;
        v8h lo = *(const v8h*)q0;
        v8h hi = *(const v8h*)(q0 + 8);
#pragma unroll
        for (int e = 0; e < 8; ++e) { Qa[e] = lo[e]; Qa[8 + e] = hi[e]; }
    }

    // Per-lane base pointers for the B-operand streams
    const _Float16* kbase = K + (size_t)col * 32 + half * 16;          // + key*32
    const _Float16* vbase = Vt + (size_t)col * NTOK + half * 16;      // + d_chunk*16*NTOK + kb

    float m[8], l[8];
    v8f Oacc[16];
#pragma unroll
    for (int r = 0; r < 8; ++r) { m[r] = -1.0e30f; l[r] = 0.0f; }
#pragma unroll
    for (int c = 0; c < 16; ++c) Oacc[c] = (v8f){};

    // Software pipeline: K tiles for block kb are loaded during block kb-32's softmax.
    v16h Kb0 = *(const v16h*)(kbase + 0 * 32);
    v16h Kb1 = *(const v16h*)(kbase + 16 * 32);

    for (int kb = 0; kb < NTOK; kb += 32) {
        v8f S0 = wmma_f16(Qa, Kb0, (v8f){});
        v8f S1 = wmma_f16(Qa, Kb1, (v8f){});

        // Issue next block's K tiles + first V tile now: their latency hides
        // under the softmax VALU section below.
        if (kb + 32 < NTOK) {
            Kb0 = *(const v16h*)(kbase + (size_t)(kb + 32) * 32);
            Kb1 = *(const v16h*)(kbase + (size_t)(kb + 48) * 32);
        }
        v16h Vcur = *(const v16h*)(vbase + kb);   // d-chunk 0

        // Online softmax. D layout: VGPR r, lane -> row = r + 8*half, col = lane&15.
        // Row reductions stay inside a 16-lane half -> shfl_xor 1,2,4,8.
        // exp() results overwrite S0/S1 in place to cap register pressure.
        v8f sv;
#pragma unroll
        for (int r = 0; r < 8; ++r) {
            float t = fmaxf(S0[r], S1[r]);
            t = fmaxf(t, __shfl_xor(t, 1, 32));
            t = fmaxf(t, __shfl_xor(t, 2, 32));
            t = fmaxf(t, __shfl_xor(t, 4, 32));
            t = fmaxf(t, __shfl_xor(t, 8, 32));
            float mn = fmaxf(m[r], t);
            float p0 = __expf(S0[r] - mn);
            float p1 = __expf(S1[r] - mn);
            float s = p0 + p1;
            s += __shfl_xor(s, 1, 32);
            s += __shfl_xor(s, 2, 32);
            s += __shfl_xor(s, 4, 32);
            s += __shfl_xor(s, 8, 32);
            float sc = __expf(m[r] - mn);
            l[r] = l[r] * sc + s;
            m[r] = mn;
            S0[r] = p0; S1[r] = p1; sv[r] = sc;
        }
#pragma unroll
        for (int c = 0; c < 16; ++c) Oacc[c] *= sv;

        // D-layout P -> LDS (row-major 16x32 f16) -> reload as A-fragment
#pragma unroll
        for (int r = 0; r < 8; ++r) {
            pb[(r + 8 * half) * 32 + col]      = (_Float16)S0[r];
            pb[(r + 8 * half) * 32 + 16 + col] = (_Float16)S1[r];
        }
        v16h Pa;
        {
            v8h lo = *(const v8h*)(pb + col * 32 + half * 16);
            v8h hi = *(const v8h*)(pb + col * 32 + half * 16 + 8);
#pragma unroll
            for (int e = 0; e < 8; ++e) { Pa[e] = lo[e]; Pa[8 + e] = hi[e]; }
        }

        // O += P @ V : 16 WMMAs with double-buffered V B-fragments so each
        // WMMA only needs a partial load wait (one load stays in flight).
#pragma unroll
        for (int c = 0; c < 15; ++c) {
            v16h Vn = *(const v16h*)(vbase + (size_t)(c + 1) * 16 * NTOK + kb);
            Oacc[c] = wmma_f16(Pa, Vcur, Oacc[c]);
            Vcur = Vn;
        }
        Oacc[15] = wmma_f16(Pa, Vcur, Oacc[15]);
    }

    // Epilogue: O/l, gamma*O + x residual (f32 out)
    v8f inv;
#pragma unroll
    for (int r = 0; r < 8; ++r) inv[r] = 1.0f / l[r];
#pragma unroll
    for (int c = 0; c < 16; ++c) {
        float g = gamma[c * 16 + col];
#pragma unroll
        for (int r = 0; r < 8; ++r) {
            size_t idx = (size_t)(tok0 + r + 8 * half) * CCH + c * 16 + col;
            out[idx] = g * (Oacc[c][r] * inv[r]) + x[idx];
        }
    }
}

extern "C" void kernel_launch(void* const* d_in, const int* in_sizes, int n_in,
                              void* d_out, int out_size, void* d_ws, size_t ws_size,
                              hipStream_t stream) {
    const float* x     = (const float*)d_in[0];
    const float* wg    = (const float*)d_in[1];
    const float* bg    = (const float*)d_in[2];
    const float* wf    = (const float*)d_in[3];
    const float* bf    = (const float*)d_in[4];
    const float* wh    = (const float*)d_in[5];
    const float* bh    = (const float*)d_in[6];
    const float* gamma = (const float*)d_in[7];
    float* out = (float*)d_out;

    char* ws = (char*)d_ws;
    _Float16* Q  = (_Float16*)(ws + 0);
    _Float16* Kb = (_Float16*)(ws + 884736);
    _Float16* Vt = (_Float16*)(ws + 1769472);
    _Float16* Pg = (_Float16*)(ws + 8847360);
    _Float16* Pf = (_Float16*)(ws + 8863744);
    _Float16* Ph = (_Float16*)(ws + 8880128);

    pack_panel<<<(8192 + 255) / 256, 256, 0, stream>>>(wg, Pg, 2, 32, 8192);
    pack_panel<<<(8192 + 255) / 256, 256, 0, stream>>>(wf, Pf, 2, 32, 8192);
    pack_panel<<<(65536 + 255) / 256, 256, 0, stream>>>(wh, Ph, 16, 256, 65536);

    // 864 16-token tiles, 4 waves (128 threads) per block
    proj_kernel<<<216, 128, 0, stream>>>(x, bg, bf, bh, Pg, Pf, Ph, Q, Kb, Vt);
    attn_kernel<<<216, 128, 0, stream>>>(x, gamma, Q, Kb, Vt, out);
}
